// CompositeLoss_54271206752628
// MI455X (gfx1250) — matrix operand order repaired and checked by reading, not compiled, not run on hardware
//
#include <hip/hip_runtime.h>
#include <stdint.h>

// Problem constants (match reference)
#define HB 8
#define HT 512
#define HD 512
#define HK 8
#define NNEG 128
#define CTEMP 0.07f

typedef __attribute__((ext_vector_type(16))) __bf16 v16bf;
typedef __attribute__((ext_vector_type(8)))  float  v8f;

#if defined(__gfx1250__) && __has_builtin(__builtin_amdgcn_global_load_async_to_lds_b128) && \
    __has_builtin(__builtin_amdgcn_s_wait_asynccnt)
#define HAVE_ASYNC_LDS 1
// The builtin takes AS(1) / AS(3) pointers to 16-byte int vectors.
typedef int v4i_ __attribute__((vector_size(16)));
typedef __attribute__((address_space(1))) v4i_ gv4i;
typedef __attribute__((address_space(3))) v4i_ lv4i;
#else
#define HAVE_ASYNC_LDS 0
#endif

#if defined(__gfx1250__) && __has_builtin(__builtin_amdgcn_tensor_load_to_lds) && \
    __has_builtin(__builtin_amdgcn_s_wait_tensorcnt)
#define HAVE_TDM 1
typedef unsigned int u32x4 __attribute__((ext_vector_type(4)));
typedef int          i32x4 __attribute__((ext_vector_type(4)));
typedef int          i32x8 __attribute__((ext_vector_type(8)));
#else
#define HAVE_TDM 0
#endif

__device__ __forceinline__ unsigned short f2bf(float f) {
    union { float f; uint32_t u; } v; v.f = f;
    uint32_t r = (v.u + 0x7FFFu + ((v.u >> 16) & 1u)) >> 16;  // RNE
    return (unsigned short)r;
}

__device__ __forceinline__ float warp_sum(float v) {
    #pragma unroll
    for (int o = 16; o > 0; o >>= 1) v += __shfl_xor(v, o, 32);
    return v;
}
__device__ __forceinline__ float warp_max(float v) {
    #pragma unroll
    for (int o = 16; o > 0; o >>= 1) v = fmaxf(v, __shfl_xor(v, o, 32));
    return v;
}

// ---------------- init scratch scalars ----------------
__global__ void k_init(float* scal) {
    if (threadIdx.x < 4) scal[threadIdx.x] = 0.0f;
}

// ---- l2-normalize rows of z into zn (fp32), raw z -> bf16 copy ----
__global__ void __launch_bounds__(256) k_norm_convert(const float* __restrict__ z,
                                                      float* __restrict__ zn,
                                                      unsigned short* __restrict__ zb) {
    int row = blockIdx.x, tid = threadIdx.x;
    const float* src = z + (size_t)row * HD;
    float v0 = src[tid], v1 = src[tid + 256];
    float ss = v0 * v0 + v1 * v1;
    __shared__ float red[8];
    __shared__ float s_inv;
    float w = warp_sum(ss);
    if ((tid & 31) == 0) red[tid >> 5] = w;
    __syncthreads();
    if (tid == 0) {
        float t = 0.f;
        #pragma unroll
        for (int i = 0; i < 8; i++) t += red[i];
        s_inv = 1.0f / fmaxf(sqrtf(t), 1e-12f);
    }
    __syncthreads();
    float inv = s_inv;
    float* dst = zn + (size_t)row * HD;
    dst[tid] = v0 * inv; dst[tid + 256] = v1 * inv;
    unsigned short* db = zb + (size_t)row * HD;
    db[tid] = f2bf(v0); db[tid + 256] = f2bf(v1);
}

// ---- in-place l2-normalize of prediction rows ----
__global__ void __launch_bounds__(256) k_norm_pred(float* __restrict__ p) {
    int row = blockIdx.x, tid = threadIdx.x;
    float* src = p + (size_t)row * HD;
    float v0 = src[tid], v1 = src[tid + 256];
    float ss = v0 * v0 + v1 * v1;
    __shared__ float red[8];
    __shared__ float s_inv;
    float w = warp_sum(ss);
    if ((tid & 31) == 0) red[tid >> 5] = w;
    __syncthreads();
    if (tid == 0) {
        float t = 0.f;
        #pragma unroll
        for (int i = 0; i < 8; i++) t += red[i];
        s_inv = 1.0f / fmaxf(sqrtf(t), 1e-12f);
    }
    __syncthreads();
    float inv = s_inv;
    src[tid] = v0 * inv; src[tid + 256] = v1 * inv;
}

// ---- convert the three D x D weight matrices to bf16 (concatenated) ----
__global__ void k_convert_w(const float* __restrict__ W1, const float* __restrict__ W5,
                            const float* __restrict__ W21, unsigned short* __restrict__ out) {
    int i = blockIdx.x * blockDim.x + threadIdx.x;
    const int MM = HD * HD;
    const float* src = (i < MM) ? W1 : ((i < 2 * MM) ? W5 : W21);
    out[i] = f2bf(src[i % MM]);
}

// ---- WMMA GEMM: zpred[m, o] = sum_d A[m, d] * W[o, d]  (bf16 in, fp32 acc) ----
// Block = 128 threads (4 waves). The 16-row A tile (bf16) is staged once into
// LDS with async global->LDS copies and shared by all 4 waves; each wave
// computes a 16x32 output strip (2 accumulators reusing one A fragment).
#define APITCH (HD + 8)   // +16B row pad to break LDS bank-conflict stride
__global__ void __launch_bounds__(128) k_gemm_wmma(const unsigned short* __restrict__ A,
                                                   const unsigned short* __restrict__ Wb,
                                                   float* __restrict__ C, int L) {
    __shared__ unsigned short sA[16 * APITCH];
    const int Mtot = HB * L;
    int tid = threadIdx.x;
    int lane = tid & 31;
    int wave = tid >> 5;
    int tileM = blockIdx.y * 16;
    int tileN = (blockIdx.x * 4 + wave) * 32;
    int half = lane >> 4;
    int r16  = lane & 15;

    // ---- stage A tile: 16 rows x 512 bf16, 16B chunks, 128 threads x 8 iters ----
    for (int i = tid; i < 16 * (HD / 8); i += 128) {
        int r = i >> 6;              // row 0..15 (64 chunks of 8 bf16 per row)
        int c = (i & 63) << 3;       // element offset within row
        int m = tileM + r; if (m >= Mtot) m = Mtot - 1;
        int b = m / L, l = m - b * L;
        const unsigned short* g = A + ((size_t)(b * HT + l)) * HD + c;
        unsigned short* s = &sA[r * APITCH + c];
#if HAVE_ASYNC_LDS
        __builtin_amdgcn_global_load_async_to_lds_b128((gv4i*)g, (lv4i*)s, 0, 0);
#else
        *(uint4*)s = *(const uint4*)g;
#endif
    }
#if HAVE_ASYNC_LDS
    __builtin_amdgcn_s_wait_asynccnt(0);
#endif
    __syncthreads();

    const unsigned short* brow0 = Wb + (size_t)(tileN + r16) * HD;
    const unsigned short* brow1 = Wb + (size_t)(tileN + 16 + r16) * HD;

    v8f acc0 = {}, acc1 = {};
    #pragma unroll 4
    for (int kb = 0; kb < HD; kb += 32) {
        if (kb + 32 < HD) {
            __builtin_prefetch((const void*)(brow0 + kb + 32), 0, 3);
            __builtin_prefetch((const void*)(brow1 + kb + 32), 0, 3);
        }
        union { v16bf v; uint32_t u[8]; } af, b0, b1;
        #pragma unroll
        for (int j = 0; j < 8; j++) {
            // ISA 16-bit A 16x32 layout: lanes 0-15: VGPR0..3 -> K0..7, VGPR4..7 -> K16..23
            //                            lanes 16-31: +8 / +8
            int k0 = (j < 4) ? (2 * j + 8 * half) : (16 + 2 * (j - 4) + 8 * half);
            af.u[j] = *(const uint32_t*)&sA[r16 * APITCH + kb + k0];   // LDS fragment read
            // B 32x16: lane = column; K in same order as A
            b0.u[j] = *(const uint32_t*)(brow0 + kb + k0);
            b1.u[j] = *(const uint32_t*)(brow1 + kb + k0);
        }
        acc0 = __builtin_amdgcn_wmma_f32_16x16x32_bf16(false, af.v, false, b0.v,
                                                       (short)0, acc0, false, false);
        acc1 = __builtin_amdgcn_wmma_f32_16x16x32_bf16(false, af.v, false, b1.v,
                                                       (short)0, acc1, false, false);
    }
    int col0 = tileN + r16;
    #pragma unroll
    for (int r = 0; r < 8; r++) {
        int mm = tileM + r + 8 * half;
        if (mm < Mtot) {
            C[(size_t)mm * HD + col0]      = acc0[r];
            C[(size_t)mm * HD + col0 + 16] = acc1[r];
        }
    }
}

// ---- CPC per-row InfoNCE: pos + 128 negatives, logsumexp CE ----
// The 2KB prediction row is staged into LDS by the Tensor Data Mover
// (single D#: 1 x 512 fp32 tile), tracked with TENSORcnt.
__global__ void __launch_bounds__(128) k_cpc_loss(const float* __restrict__ pred,
                                                  const float* __restrict__ zn,
                                                  const int* __restrict__ nidx,
                                                  int L, int horizon, float weight,
                                                  float* __restrict__ acc) {
    int n = blockIdx.x;       // 0 .. B*L-1
    int tid = threadIdx.x;    // 128
    __shared__ float spred[HD];
    __shared__ float lneg[NNEG];
    __shared__ float red[4];
    __shared__ float s_pos, s_max;

    const float* pr = pred + (size_t)n * HD;
#if HAVE_TDM
    if (tid < 32) {  // wave 0 issues one TDM descriptor (EXEC ignored by TDM)
        unsigned lds_off = (unsigned)(size_t)(__attribute__((address_space(3))) float*)spred;
        unsigned long long ga = (unsigned long long)(size_t)pr;
        u32x4 g0;
        g0.x = 1u;                                                  // count=1 user D#
        g0.y = lds_off;                                             // LDS byte address
        g0.z = (unsigned)(ga & 0xFFFFFFFFu);                        // global_addr[31:0]
        g0.w = (unsigned)((ga >> 32) & 0x01FFFFFFu) | (2u << 30);   // addr[56:32] | type=2
        i32x8 g1;
        g1[0] = 0x20000;                  // data_size=4B; no multicast/pad/iterate
        g1[1] = (int)(512u << 16);        // tensor_dim0 = 512 (bits 79:48)
        g1[2] = (int)(1u << 16);          // tensor_dim1 = 1   (bits 111:80)
        g1[3] = (int)(512u << 16);        // tile_dim0 = 512   (bits 127:112)
        g1[4] = 1;                        // tile_dim1 = 1, tile_dim2 = 0
        g1[5] = 512;                      // tensor_dim0_stride = 512
        g1[6] = 0;
        g1[7] = 0;
        i32x4 zz = {0, 0, 0, 0};
#if __clang_major__ >= 23
        i32x8 z8 = {0, 0, 0, 0, 0, 0, 0, 0};
        __builtin_amdgcn_tensor_load_to_lds(g0, g1, zz, zz, z8, 0);
#else
        __builtin_amdgcn_tensor_load_to_lds(g0, g1, zz, zz, 0);
#endif
        __builtin_amdgcn_s_wait_tensorcnt(0);
    }
    __syncthreads();
#else
    for (int d = tid; d < HD; d += 128) spred[d] = pr[d];
    __syncthreads();
#endif

    int b = n / L, l = n - b * L;
    const float* zpos = zn + (size_t)(b * HT + l + horizon) * HD;
    float p = 0.f;
    for (int d = tid; d < HD; d += 128) p += spred[d] * zpos[d];
    p = warp_sum(p);
    if ((tid & 31) == 0) red[tid >> 5] = p;
    __syncthreads();
    if (tid == 0) s_pos = (red[0] + red[1] + red[2] + red[3]) / CTEMP;

    int idx = nidx[(size_t)n * NNEG + tid];
    const float* zneg = zn + (size_t)idx * HD;
    float dn = 0.f;
    for (int d = 0; d < HD; d++) dn += spred[d] * zneg[d];
    lneg[tid] = dn / CTEMP;
    __syncthreads();

    float mv = warp_max(lneg[tid]);
    if ((tid & 31) == 0) red[tid >> 5] = mv;
    __syncthreads();
    if (tid == 0) {
        float mm = fmaxf(fmaxf(red[0], red[1]), fmaxf(red[2], red[3]));
        s_max = fmaxf(mm, s_pos);
    }
    __syncthreads();
    float e = expf(lneg[tid] - s_max);
    e = warp_sum(e);
    __syncthreads();
    if ((tid & 31) == 0) red[tid >> 5] = e;
    __syncthreads();
    if (tid == 0) {
        float ssum = red[0] + red[1] + red[2] + red[3] + expf(s_pos - s_max);
        float lse = s_max + logf(ssum);
        float ce = lse - s_pos;
        atomicAdd(acc, weight * ce);
    }
}

// ---- per-dimension variance (ddof=1), accumulate mean into acc ----
__global__ void __launch_bounds__(256) k_variance(const float* __restrict__ z,
                                                  float* __restrict__ acc) {
    int d = blockIdx.x, tid = threadIdx.x;
    float s = 0.f, q = 0.f;
    for (int r = tid; r < HB * HT; r += 256) {
        float v = z[(size_t)r * HD + d];
        s += v; q += v * v;
    }
    __shared__ float rs[8], rq[8];
    float ws_ = warp_sum(s), wq = warp_sum(q);
    if ((tid & 31) == 0) { rs[tid >> 5] = ws_; rq[tid >> 5] = wq; }
    __syncthreads();
    if (tid == 0) {
        float S = 0.f, Q = 0.f;
        #pragma unroll
        for (int i = 0; i < 8; i++) { S += rs[i]; Q += rq[i]; }
        const float nf = (float)(HB * HT);
        float var = (Q - S * S / nf) / (nf - 1.0f);
        atomicAdd(acc, var / (float)HD);
    }
}

// ---- Gaussian diag log-emission: wave s handles state s for one (b,t) ----
__global__ void __launch_bounds__(256) k_hmm_emit(const float* __restrict__ z,
                                                  const float* __restrict__ means,
                                                  const float* __restrict__ logvars,
                                                  float* __restrict__ logemit) {
    int bt = blockIdx.x;
    int lane = threadIdx.x & 31;
    int s = threadIdx.x >> 5;
    const float* zr = z + (size_t)bt * HD;
    const float* mu = means + (size_t)s * HD;
    const float* lv = logvars + (size_t)s * HD;
    float quad = 0.f, ldet = 0.f;
    for (int d = lane; d < HD; d += 32) {
        float var = fmaxf(expf(lv[d]), 1e-4f);
        float df = zr[d] - mu[d];
        quad += df * df / var;
        ldet += logf(var);
    }
    quad = warp_sum(quad);
    ldet = warp_sum(ldet);
    if (lane == 0) {
        const float c = (float)HD * 1.8378770664093453f;  // D * log(2*pi)
        logemit[(size_t)bt * HK + s] = -0.5f * (c + ldet + quad);
    }
}

// ---- HMM forward scan (single block, 64 = B*K threads) ----
__global__ void __launch_bounds__(64) k_hmm_forward(const float* __restrict__ logemit,
                                                    const float* __restrict__ init_logits,
                                                    const float* __restrict__ trans,
                                                    float* __restrict__ out) {
    __shared__ float logA[HK * HK];
    __shared__ float logpi[HK];
    __shared__ float alpha[HB * HK];
    int tid = threadIdx.x;
    int b = tid >> 3, s = tid & 7;

    if (tid < HK) {
        float mx = -1e30f;
        for (int j = 0; j < HK; j++) mx = fmaxf(mx, trans[tid * HK + j]);
        float sm = 0.f;
        for (int j = 0; j < HK; j++) sm += expf(trans[tid * HK + j] - mx);
        float lse = mx + logf(sm);
        for (int j = 0; j < HK; j++) logA[tid * HK + j] = trans[tid * HK + j] - lse;
    }
    if (tid == 0) {
        float mx = -1e30f;
        for (int j = 0; j < HK; j++) mx = fmaxf(mx, init_logits[j]);
        float sm = 0.f;
        for (int j = 0; j < HK; j++) sm += expf(init_logits[j] - mx);
        float lse = mx + logf(sm);
        for (int j = 0; j < HK; j++) logpi[j] = init_logits[j] - lse;
    }
    __syncthreads();
    alpha[tid] = logpi[s] + logemit[((size_t)b * HT) * HK + s];
    __syncthreads();
    for (int t = 1; t < HT; t++) {
        float mx = -1e30f;
        #pragma unroll
        for (int i = 0; i < HK; i++) mx = fmaxf(mx, alpha[b * HK + i] + logA[i * HK + s]);
        float sm = 0.f;
        #pragma unroll
        for (int i = 0; i < HK; i++) sm += expf(alpha[b * HK + i] + logA[i * HK + s] - mx);
        float v = mx + logf(sm) + logemit[((size_t)b * HT + t) * HK + s];
        __syncthreads();
        alpha[tid] = v;
        __syncthreads();
    }
    if (tid == 0) {
        float tot = 0.f;
        for (int bb = 0; bb < HB; bb++) {
            float mx = -1e30f;
            for (int i = 0; i < HK; i++) mx = fmaxf(mx, alpha[bb * HK + i]);
            float sm = 0.f;
            for (int i = 0; i < HK; i++) sm += expf(alpha[bb * HK + i] - mx);
            tot += mx + logf(sm);
        }
        *out = -tot / (float)HB;
    }
}

// ---- combine: out = cpc + hmm + (-log(var_mean + 1e-6)) ----
__global__ void k_final(const float* __restrict__ scal, float* __restrict__ out) {
    out[0] = scal[0] + scal[2] + (-logf(scal[1] + 1e-6f));
}

extern "C" void kernel_launch(void* const* d_in, const int* in_sizes, int n_in,
                              void* d_out, int out_size, void* d_ws, size_t ws_size,
                              hipStream_t stream) {
    const float* z           = (const float*)d_in[0];
    const float* W1          = (const float*)d_in[1];
    const float* W5          = (const float*)d_in[2];
    const float* W21         = (const float*)d_in[3];
    const float* init_logits = (const float*)d_in[4];
    const float* trans       = (const float*)d_in[5];
    const float* means       = (const float*)d_in[6];
    const float* logvars     = (const float*)d_in[7];
    const int* nidx[3] = { (const int*)d_in[8], (const int*)d_in[9], (const int*)d_in[10] };

    char* ws = (char*)d_ws;
    float*          zn      = (float*)(ws);                          //  8 MB  fp32 normalized all_z
    unsigned short* zb      = (unsigned short*)(ws + (8u  << 20));   //  4 MB  bf16 raw z
    unsigned short* Wb      = (unsigned short*)(ws + (12u << 20));   // 1.5 MB bf16 W1|W5|W21
    float*          zpred   = (float*)(ws + (14u << 20));            //  8 MB  predictions
    float*          logemit = (float*)(ws + (22u << 20));            // 128 KB
    float*          scal    = (float*)(ws + (23u << 20));            // accumulators

    k_init<<<1, 64, 0, stream>>>(scal);
    k_norm_convert<<<HB * HT, 256, 0, stream>>>(z, zn, zb);
    k_convert_w<<<(3 * HD * HD) / 256, 256, 0, stream>>>(W1, W5, W21, Wb);

    const int   hor[3] = { 1, 5, 21 };
    const float rw[3]  = { 1.0f, 0.44721359549995793f, 0.21821789023599239f };
    const float tot    = rw[0] + rw[1] + rw[2];

    for (int i = 0; i < 3; i++) {
        int L = HT - hor[i];
        int M = HB * L;
        // each block covers 16 rows x 128 cols (4 waves x 32 cols)
        dim3 grid(HD / 128, (M + 15) / 16);
        k_gemm_wmma<<<grid, 128, 0, stream>>>(zb, Wb + (size_t)i * HD * HD, zpred, L);
        k_norm_pred<<<M, 256, 0, stream>>>(zpred);
        float w = (rw[i] / tot) / (float)M;
        k_cpc_loss<<<M, 128, 0, stream>>>(zpred, zn, nidx[i], L, hor[i], w, &scal[0]);
    }

    k_variance<<<HD, 256, 0, stream>>>(z, &scal[1]);
    k_hmm_emit<<<HB * HT, 256, 0, stream>>>(z, means, logvars, logemit);
    k_hmm_forward<<<1, 64, 0, stream>>>(logemit, init_logits, trans, &scal[2]);
    k_final<<<1, 1, 0, stream>>>(scal, (float*)d_out);
}
